// FarGANWaveHeadLite_22565758173234
// MI455X (gfx1250) — matrix operand order repaired and checked by reading, not compiled, not run on hardware
//
#include <hip/hip_runtime.h>
#include <hip/hip_bf16.h>
#include <math.h>

// ---------------------------------------------------------------------------
// FarGAN wave head for MI455X (gfx1250, wave32, WMMA f16 16x16x32)
//
//  * batch B=16 == WMMA M-tile -> ONE persistent workgroup runs the whole
//    8000-step GRU scan with full batch in the M dimension.
//  * 18 waves x 2 column tiles cover the 576-wide gate outputs per layer.
//  * f16 weights pre-packed in B-fragment lane order, streamed from L2
//    with inline global_load_b128; unroll-2 k-loops with EVEN trip counts
//    (layer-0 K padded 194->256) so no instance gets flattened/spilled.
//  * hidden state / x / gate sums in LDS (~52KB), f32 elementwise update.
// ---------------------------------------------------------------------------

typedef __attribute__((ext_vector_type(16))) _Float16 v16h;
typedef __attribute__((ext_vector_type(8)))  float    v8f;

#define B_     16
#define T_     50
#define SF_    4
#define SPS_   40
#define S_     8000
#define DIN_   48
#define PED_   64
#define COND_  128
#define GRUD_  192
#define G3_    576
#define NT_    36          // 576/16 column tiles
#define KC_IH0 8           // 194 -> padded to 256 = 8 chunks of K=32 (even!)
#define KC_HH  6           // 192 = 6 chunks of K=32
#define XSTR_  256         // xL row stride (halves)
#define TWO_PI_F 6.2831853071795864f

// packed weight sizes in halves: per (tile,chunk): 32 lanes * 16 halves = 512
#define PK_IH0 (NT_*KC_IH0*512)   // 147456
#define PK_STD (NT_*KC_HH*512)    // 110592

// ---------------- WMMA helpers -------------------------------------------

__device__ __forceinline__ v8f wmma_f16(v16h a, v16h b, v8f c) {
  return __builtin_amdgcn_wmma_f32_16x16x32_f16(
      /*neg_a=*/false, a, /*neg_b=*/false, b,
      /*c_mod=*/(short)0, c, /*reuse_a=*/false, /*reuse_b=*/false);
}

// A fragment (16x32 f16, M=row striped per lane) from LDS row-major [m][k]
// layout per ISA: lanes 0-15: K = 0..7,16..23 ; lanes 16-31: K = 8..15,24..31
__device__ __forceinline__ v16h load_a_frag(const _Float16* base, int rs, int k0) {
  int lane = threadIdx.x & 31;
  int m    = lane & 15;
  int koff = (lane >> 4) << 3;           // 0 or 8
  const _Float16* p = base + m * rs + k0 + koff;
  union { v16h v; float4 f[2]; } u;
  u.f[0] = *(const float4*)(p);          // K = k0+koff   .. +7
  u.f[1] = *(const float4*)(p + 16);     // K = k0+koff+16 .. +23
  return u.v;
}

// B fragment (32x16 f16, N=col striped per lane) from packed global weights.
// Packed so each lane reads 16 contiguous halves (2x global_load_b128).
__device__ __forceinline__ v16h load_b_frag(const _Float16* base, int tc) {
  int lane = threadIdx.x & 31;
  const _Float16* p = base + ((long)tc * 32 + lane) * 16;
  union { v16h v; float4 f[2]; } u;
  u.f[0] = *(const float4*)(p);
  u.f[1] = *(const float4*)(p + 8);
  return u.v;
}

// ---------------- per-layer GRU step (matmuls + gates + update) ------------
// KC compile-time & even; unroll-2 keeps live fragments small (no spills).
template <int KC>
__device__ __forceinline__ void gru_layer_step(
    const _Float16* __restrict__ asrc, const int rs,   // gi input (LDS)
    _Float16* __restrict__ h,                           // hL[l]  (LDS)
    const _Float16* __restrict__ wih,                   // packed W_ih (global)
    const _Float16* __restrict__ whh,                   // packed W_hh (global)
    const float* __restrict__ bih, const float* __restrict__ bhh,
    _Float16* __restrict__ g_rz, _Float16* __restrict__ g_in,
    _Float16* __restrict__ g_hn,
    const int tid, const int wv, const int nloc, const int mbase,
    const int t0, const int t1) {
  v8f gi0 = {0,0,0,0,0,0,0,0}, gi1 = {0,0,0,0,0,0,0,0};
  v8f gh0 = {0,0,0,0,0,0,0,0}, gh1 = {0,0,0,0,0,0,0,0};
  #pragma unroll 2
  for (int c = 0; c < KC; ++c) {                 // gi = x @ Wih^T
    v16h a  = load_a_frag(asrc, rs, c * 32);
    v16h b0 = load_b_frag(wih, t0 * KC + c);
    v16h b1 = load_b_frag(wih, t1 * KC + c);
    gi0 = wmma_f16(a, b0, gi0);
    gi1 = wmma_f16(a, b1, gi1);
  }
  #pragma unroll 2
  for (int c = 0; c < KC_HH; ++c) {              // gh = h @ Whh^T
    v16h a  = load_a_frag(h, GRUD_, c * 32);
    v16h b0 = load_b_frag(whh, t0 * KC_HH + c);
    v16h b1 = load_b_frag(whh, t1 * KC_HH + c);
    gh0 = wmma_f16(a, b0, gh0);
    gh1 = wmma_f16(a, b1, gh1);
  }
  const int n0 = t0 * 16 + nloc, n1 = t1 * 16 + nloc;  // global gate cols
  if (wv < 12) {                                  // r (0..191) | z (192..383)
    float bb0 = bih[n0] + bhh[n0], bb1 = bih[n1] + bhh[n1];
    #pragma unroll
    for (int v = 0; v < 8; ++v) {
      int m = mbase + v;
      g_rz[m * 384 + n0] = (_Float16)(gi0[v] + gh0[v] + bb0);
      g_rz[m * 384 + n1] = (_Float16)(gi1[v] + gh1[v] + bb1);
    }
  } else {                                        // n gate (384..575)
    int l0 = n0 - 384, l1 = n1 - 384;
    float bi0 = bih[n0], bi1 = bih[n1], bh0 = bhh[n0], bh1 = bhh[n1];
    #pragma unroll
    for (int v = 0; v < 8; ++v) {
      int m = mbase + v;
      g_in[m * GRUD_ + l0] = (_Float16)(gi0[v] + bi0);
      g_in[m * GRUD_ + l1] = (_Float16)(gi1[v] + bi1);
      g_hn[m * GRUD_ + l0] = (_Float16)(gh0[v] + bh0);
      g_hn[m * GRUD_ + l1] = (_Float16)(gh1[v] + bh1);
    }
  }
  __syncthreads();
  // elementwise GRU update (f32)
  for (int idx = tid; idx < B_ * GRUD_; idx += 576) {
    int m = idx / GRUD_, j = idx - m * GRUD_;
    float r  = 1.f / (1.f + __expf(-(float)g_rz[m * 384 + j]));
    float z  = 1.f / (1.f + __expf(-(float)g_rz[m * 384 + 192 + j]));
    float nn = tanhf((float)g_in[idx] + r * (float)g_hn[idx]);
    float hp = (float)h[idx];
    h[idx] = (_Float16)((1.f - z) * nn + z * hp);
  }
  __syncthreads();
}

// ---------------- weight packing (f32 row-major -> f16 B-fragment order) ---
// B-layout: lane n=L%16 is the column; halves j=0..15 -> k = (j&7) + (L>=16 ? 8:0) + (j>=8 ? 16:0)
__global__ void pack_w_kernel(const float* __restrict__ src,
                              _Float16* __restrict__ dst, int K, int KC) {
  int i = blockIdx.x * blockDim.x + threadIdx.x;
  int total = NT_ * KC * 512;
  if (i >= total) return;
  int j    = i & 15;
  int lane = (i >> 4) & 31;
  int tc   = i >> 9;
  int c    = tc % KC;
  int tile = tc / KC;
  int n = tile * 16 + (lane & 15);
  int k = c * 32 + (j & 7) + ((lane >= 16) ? 8 : 0) + ((j >> 3) << 4);
  float v = (k < K) ? src[n * K + k] : 0.f;
  dst[i] = (_Float16)v;
}

// ---------------- frame-rate conditioning network --------------------------
__global__ __launch_bounds__(512) void frame_cond_kernel(
    const float* __restrict__ feats, const float* __restrict__ pemb,
    const float* __restrict__ w1, const float* __restrict__ b1,
    const float* __restrict__ w2, const float* __restrict__ b2,
    _Float16* __restrict__ pe_h, _Float16* __restrict__ cond_h,
    float* __restrict__ dvbuf, float* __restrict__ vbuf) {
  __shared__ float fin[112];
  __shared__ float h1[256];
  __shared__ int   sidx;
  int bt  = blockIdx.x;            // b*50 + t
  int tid = threadIdx.x;
  if (tid < DIN_) fin[tid] = feats[bt * DIN_ + tid];
  __syncthreads();
  if (tid == 0) {
    float f0 = fin[46], vo = fin[47];
    float period = 256.f * exp2f(-(f0 + 2.f));
    period = fminf(fmaxf(period, 32.f), 255.f);
    int idx = (int)roundf(period);
    idx = (idx < 32 ? 32 : (idx > 255 ? 255 : idx)) - 32;
    sidx = idx;
    float f0hz  = 16000.f / fmaxf(period, 1.f);
    float v     = fminf(fmaxf(vo, 0.f), 1.f);
    float delta = TWO_PI_F * fmaxf(f0hz, 60.f) / 16000.f;
    dvbuf[bt] = delta * v;
    vbuf[bt]  = v;
  }
  __syncthreads();
  if (tid < PED_) {
    float p = pemb[sidx * PED_ + tid];
    fin[DIN_ + tid]        = p;
    pe_h[bt * PED_ + tid]  = (_Float16)p;
  }
  __syncthreads();
  if (tid < 256) {
    float acc = b1[tid];
    for (int j = 0; j < 112; ++j) acc += w1[tid * 112 + j] * fin[j];
    h1[tid] = 0.5f * acc * (1.f + erff(acc * 0.70710678118f));  // exact GELU
  }
  __syncthreads();
  {
    float acc = b2[tid];
    for (int j = 0; j < 256; ++j) acc += w2[tid * 256 + j] * h1[j];
    cond_h[bt * 512 + tid] = (_Float16)acc;   // (b,t,sf,cond) row-major
  }
}

// ---------------- phase cumsum ---------------------------------------------
__global__ void phase_kernel(const float* __restrict__ dvbuf,
                             float* __restrict__ ps, float* __restrict__ dvsf) {
  int b = threadIdx.x;
  if (b >= B_) return;
  float cum = 0.f;
  for (int i = 0; i < 200; ++i) {
    float dv = dvbuf[b * T_ + (i >> 2)];
    ps[b * 200 + i]   = fmodf(cum, TWO_PI_F);  // exclusive prefix of inc
    dvsf[b * 200 + i] = dv;
    cum += dv * (float)SPS_;
  }
}

// ---------------- persistent GRU scan (the WMMA kernel) --------------------
__global__ __launch_bounds__(576, 1) void gru_persistent_kernel(
    const _Float16* __restrict__ cond_h, const _Float16* __restrict__ pe_h,
    const float* __restrict__ ps, const float* __restrict__ dvsf,
    const float* __restrict__ vbuf,
    const _Float16* __restrict__ wpack,
    const float* __restrict__ bih0, const float* __restrict__ bhh0,
    const float* __restrict__ bih1, const float* __restrict__ bhh1,
    const float* __restrict__ bih2, const float* __restrict__ bhh2,
    const float* __restrict__ outw, const float* __restrict__ outb,
    const float* __restrict__ lgain, const float* __restrict__ noise,
    float* __restrict__ out) {
  __shared__ __align__(16) _Float16 xL[B_ * XSTR_];        // layer0 input, K padded
  __shared__ __align__(16) _Float16 h0[B_ * GRUD_];
  __shared__ __align__(16) _Float16 h1[B_ * GRUD_];
  __shared__ __align__(16) _Float16 h2[B_ * GRUD_];
  __shared__ __align__(16) _Float16 g_rz[B_ * 384];        // r|z gate sums
  __shared__ __align__(16) _Float16 g_in[B_ * GRUD_];      // i_n
  __shared__ __align__(16) _Float16 g_hn[B_ * GRUD_];      // h_n
  __shared__ float partial[B_ * 16];

  const int tid  = threadIdx.x;
  const int wv   = tid >> 5;            // wave 0..17
  const int lane = tid & 31;
  const int nloc = lane & 15;
  const int mbase = (lane >> 4) * 8;    // C-frag: rows m = mbase + vgpr
  const int t0 = 2 * wv, t1 = 2 * wv + 1;

  for (int i = tid; i < B_ * XSTR_; i += 576) xL[i] = (_Float16)0.f;
  for (int i = tid; i < B_ * GRUD_; i += 576) {
    h0[i] = (_Float16)0.f; h1[i] = (_Float16)0.f; h2[i] = (_Float16)0.f;
  }
  __syncthreads();

  const float out_bias = outb[0];
  const float gain = fminf(fmaxf(expf(lgain[0]), 0.5f), 1.5f);

  for (int s = 0; s < S_; ++s) {
    const int g  = s / SPS_;             // subframe 0..199
    const int t  = s / (SPS_ * SF_);     // frame 0..49
    const int sf = (s / SPS_) & 3;

    // --- refresh x: cond+pe once per subframe, sin/cos every sample ---
    if ((s % SPS_) == 0) {
      for (int idx = tid; idx < B_ * GRUD_; idx += 576) {
        int m = idx / GRUD_, q = idx - m * GRUD_;
        _Float16 val = (q < COND_)
            ? cond_h[((m * T_ + t) * SF_ + sf) * COND_ + q]
            : pe_h[(m * T_ + t) * PED_ + (q - COND_)];
        xL[m * XSTR_ + q] = val;
      }
    }
    if (tid < 32) {
      int m = tid >> 1;
      float ph = ps[m * 200 + g] + dvsf[m * 200 + g] * (float)(s - g * SPS_);
      xL[m * XSTR_ + 192 + (tid & 1)] =
          (_Float16)((tid & 1) ? __cosf(ph) : __sinf(ph));
    }
    __syncthreads();

    // --- three GRU layers; weight ptrs derive directly from the kernel
    //     argument so loads lower to global_load_b128 ---
    gru_layer_step<KC_IH0>(xL, XSTR_, h0,
                           wpack, wpack + PK_IH0,
                           bih0, bhh0, g_rz, g_in, g_hn,
                           tid, wv, nloc, mbase, t0, t1);
    gru_layer_step<KC_HH>(h0, GRUD_, h1,
                          wpack + PK_IH0 + 1 * PK_STD,
                          wpack + PK_IH0 + 2 * PK_STD,
                          bih1, bhh1, g_rz, g_in, g_hn,
                          tid, wv, nloc, mbase, t0, t1);
    gru_layer_step<KC_HH>(h1, GRUD_, h2,
                          wpack + PK_IH0 + 3 * PK_STD,
                          wpack + PK_IH0 + 4 * PK_STD,
                          bih2, bhh2, g_rz, g_in, g_hn,
                          tid, wv, nloc, mbase, t0, t1);

    // --- output head: sample = outw . h2 + b ---
    if (tid < 256) {
      int m = tid >> 4, seg = tid & 15;
      int jb = seg * 12;
      float acc = 0.f;
      #pragma unroll
      for (int j = 0; j < 12; ++j)
        acc += outw[jb + j] * (float)h2[m * GRUD_ + jb + j];
      partial[m * 16 + seg] = acc;
    }
    __syncthreads();
    if (tid < B_) {
      int m = tid;
      float sample = out_bias;
      #pragma unroll
      for (int k = 0; k < 16; ++k) sample += partial[m * 16 + k];
      float vg    = vbuf[m * T_ + t];
      float nz    = noise[m * S_ + s] * 0.003f;
      float noisy = 0.6f * sample + 0.4f * nz;
      float smp   = vg * sample + (1.f - vg) * noisy;
      float w     = fminf(fmaxf(smp, -1.f), 1.f);
      out[m * S_ + s] = 1.1f * tanhf(0.9f * gain * w);
    }
    __syncthreads();
  }
}

// ---------------- launch ----------------------------------------------------
extern "C" void kernel_launch(void* const* d_in, const int* in_sizes, int n_in,
                              void* d_out, int out_size, void* d_ws, size_t ws_size,
                              hipStream_t stream) {
  const float* feats = (const float*)d_in[0];
  const float* noise = (const float*)d_in[1];
  const float* pemb  = (const float*)d_in[2];
  const float* w1    = (const float*)d_in[3];
  const float* b1    = (const float*)d_in[4];
  const float* w2    = (const float*)d_in[5];
  const float* b2    = (const float*)d_in[6];
  const float* outw  = (const float*)d_in[7];
  const float* outb  = (const float*)d_in[8];
  const float* lgain = (const float*)d_in[9];
  const float* srcs[6] = { (const float*)d_in[10], (const float*)d_in[11],   // ih0, hh0
                           (const float*)d_in[14], (const float*)d_in[15],   // ih1, hh1
                           (const float*)d_in[18], (const float*)d_in[19] }; // ih2, hh2
  const float* bih[3] = { (const float*)d_in[12], (const float*)d_in[16], (const float*)d_in[20] };
  const float* bhh[3] = { (const float*)d_in[13], (const float*)d_in[17], (const float*)d_in[21] };
  float* out = (float*)d_out;

  // workspace layout (halves then floats), ~2.4 MB total
  _Float16* wpack = (_Float16*)d_ws;
  size_t off = (size_t)PK_IH0 + 5u * (size_t)PK_STD;          // 700416 halves
  _Float16* pe_h   = wpack + off; off += (size_t)B_ * T_ * PED_;          // 51200
  _Float16* cond_h = wpack + off; off += (size_t)B_ * T_ * SF_ * COND_;   // 409600
  float* fws   = (float*)(wpack + off);
  float* dvbuf = fws;            // 800
  float* vbuf  = fws + 800;      // 800
  float* psb   = fws + 1600;     // 3200
  float* dvsf  = fws + 4800;     // 3200

  _Float16* dsts[6] = { wpack, wpack + PK_IH0,
                        wpack + PK_IH0 + 1 * PK_STD, wpack + PK_IH0 + 2 * PK_STD,
                        wpack + PK_IH0 + 3 * PK_STD, wpack + PK_IH0 + 4 * PK_STD };
  int KCs[6] = { KC_IH0, KC_HH, KC_HH, KC_HH, KC_HH, KC_HH };
  int Ks[6]  = { 194, 192, 192, 192, 192, 192 };
  for (int m = 0; m < 6; ++m) {
    int total = NT_ * KCs[m] * 512;
    pack_w_kernel<<<(total + 255) / 256, 256, 0, stream>>>(srcs[m], dsts[m], Ks[m], KCs[m]);
  }
  frame_cond_kernel<<<800, 512, 0, stream>>>(feats, pemb, w1, b1, w2, b2,
                                             pe_h, cond_h, dvbuf, vbuf);
  phase_kernel<<<1, 32, 0, stream>>>(dvbuf, psb, dvsf);
  gru_persistent_kernel<<<1, 576, 0, stream>>>(
      cond_h, pe_h, psb, dvsf, vbuf, wpack,
      bih[0], bhh[0], bih[1], bhh[1], bih[2], bhh[2],
      outw, outb, lgain, noise, out);
}
